// AttentionBasedDecoder_78769700208942
// MI455X (gfx1250) — matrix operand order repaired and checked by reading, not compiled, not run on hardware
//
#include <hip/hip_runtime.h>
#include <math.h>

// ---------------------------------------------------------------------------
// Problem constants (from reference): B=32, F=512, D=256, T=32, V=128
// ---------------------------------------------------------------------------
#define Bc 32
#define Fc 512
#define Dc 256
#define Tc 32
#define BFc (Bc * Fc)

// ---------------------------------------------------------------------------
// WMMA types (CDNA5 / gfx1250, wave32)
// ---------------------------------------------------------------------------
typedef __attribute__((ext_vector_type(16))) __bf16        v16bf;
typedef __attribute__((ext_vector_type(8)))  float         v8f;
typedef __attribute__((ext_vector_type(4)))  unsigned int  u32x4;

union Frag {
    v16bf bf;
    u32x4 u[2];
};

__device__ __forceinline__ unsigned short f2bf(float x) {
    unsigned int u = __float_as_uint(x);
    u = u + 0x7FFFu + ((u >> 16) & 1u);   // round-to-nearest-even
    return (unsigned short)(u >> 16);
}
__device__ __forceinline__ float sigf(float x) { return 1.0f / (1.0f + __expf(-x)); }

// ---------------------------------------------------------------------------
// Per-wave GEMM: out16x256 = A[16,256](bf16, LDS) x W^T  (W row-major [256,256] bf16)
// Each wave (of 8) computes two 16x16 N-tiles, K swept in 8 chunks of 32.
// A-fragment (16-bit A 16x32, ISA 7.12.2): lane m=lane&15, g=lane>>4;
//   halves p=0..7 -> K = k0 + p + 8g  (one b128), p=8..15 -> K = k0+16+(p-8)+8g (one b128)
// B-fragment (16-bit B 32x16): lane n=lane&15, g=lane>>4; same K mapping as A-fragment
// D-fragment: lane n=lane&15, g=lane>>4; VGPR r -> M = r + 8g
// ---------------------------------------------------------------------------
__device__ __forceinline__ void wave_gemm16(const unsigned short* __restrict__ As,   // LDS [16*256] bf16
                                            const unsigned short* __restrict__ W,    // global [256*256] bf16
                                            float* __restrict__ Es,                  // LDS [16*256] f32
                                            int lane, int wave) {
    const int m = lane & 15;
    const int g = lane >> 4;
    const int n = lane & 15;
#pragma unroll
    for (int ti = 0; ti < 2; ++ti) {
        const int nt = wave * 2 + ti;
        Frag a, b;
        v8f acc = {};
#pragma unroll
        for (int kk = 0; kk < 8; ++kk) {
            const int k0 = kk * 32;
            a.u[0] = *(const u32x4*)&As[m * Dc + k0 + 8 * g];
            a.u[1] = *(const u32x4*)&As[m * Dc + k0 + 16 + 8 * g];
            const unsigned short* wr = &W[(nt * 16 + n) * Dc + k0];
            b.u[0] = *(const u32x4*)&wr[8 * g];
            b.u[1] = *(const u32x4*)&wr[16 + 8 * g];
            acc = __builtin_amdgcn_wmma_f32_16x16x32_bf16(
                false, a.bf, false, b.bf, (short)0, acc, false, false);
        }
#pragma unroll
        for (int r = 0; r < 8; ++r)
            Es[(r + 8 * g) * Dc + nt * 16 + n] = acc[r];
    }
}

// ---------------------------------------------------------------------------
// Precompute kernels
// ---------------------------------------------------------------------------
__global__ void pe_kernel(float* __restrict__ q) {                 // q[F,D] position encoding
    int idx = blockIdx.x * blockDim.x + threadIdx.x;
    if (idx >= Fc * Dc) return;
    int pos = idx / Dc, c = idx % Dc;
    float i2  = (float)((c / 2) * 2);
    float ang = (float)pos / powf(10000.0f, i2 / (float)Dc);
    q[idx] = (c & 1) ? cosf(ang) : sinf(ang);
}

__global__ void dec_gather_kernel(const int* __restrict__ ids, const float* __restrict__ emb,
                                  float* __restrict__ dec) {        // dec[B,T,D] = emb[ids]
    int idx = blockIdx.x * blockDim.x + threadIdx.x;
    if (idx >= Bc * Tc * Dc) return;
    int d = idx % Dc, bt = idx / Dc;
    dec[idx] = emb[ids[bt] * Dc + d];
}

__global__ void cvt_bf16_kernel(const float* __restrict__ src, unsigned short* __restrict__ dst, int n) {
    int idx = blockIdx.x * blockDim.x + threadIdx.x;
    if (idx < n) dst[idx] = f2bf(src[idx]);
}

__global__ void qwq_kernel(const float* __restrict__ q, const float* __restrict__ w_q,
                           float* __restrict__ qwq) {               // qwq[f,o] = q[f,:] . w_q[o,:]
    int idx = blockIdx.x * blockDim.x + threadIdx.x;
    if (idx >= Fc * Dc) return;
    int fr = idx / Dc, o = idx % Dc;
    float s = 0.f;
    for (int k = 0; k < Dc; ++k) s += q[fr * Dc + k] * w_q[o * Dc + k];
    qwq[idx] = s;
}

__global__ void fq_kernel(const float* __restrict__ f, const float* __restrict__ q,
                          float* __restrict__ fq) {                 // fq = f + q (broadcast over B)
    int idx = blockIdx.x * blockDim.x + threadIdx.x;
    if (idx >= Bc * Fc * Dc) return;
    fq[idx] = f[idx] + q[idx % (Fc * Dc)];
}

// base[m,o] = f[m,:] @ w_f^T + qwq[m%F, o]   via WMMA
__global__ __launch_bounds__(256)
void base_kernel(const float* __restrict__ f, const unsigned short* __restrict__ wfb,
                 const float* __restrict__ qwq, float* __restrict__ base) {
    __shared__ __align__(16) unsigned short As[16 * Dc];
    __shared__ float Es[16 * Dc];
    const int tid = threadIdx.x;
    const int rowbase = blockIdx.x * 16;
    for (int i = tid; i < 16 * Dc; i += 256) As[i] = f2bf(f[rowbase * Dc + i]);
    __syncthreads();
    wave_gemm16(As, wfb, Es, tid & 31, tid >> 5);
    __syncthreads();
    for (int i = tid; i < 16 * Dc; i += 256) {
        int col = i & (Dc - 1);
        int row = i >> 8;
        int frow = (rowbase + row) & (Fc - 1);
        base[rowbase * Dc + i] = Es[i] + qwq[frow * Dc + col];
    }
}

// ---------------------------------------------------------------------------
// Per-step small kernels
// ---------------------------------------------------------------------------
__global__ void gates_kernel(const float* __restrict__ dec, const float* __restrict__ cp,
                             const float* __restrict__ h, const float* __restrict__ w_ih,
                             const float* __restrict__ w_hh, float* __restrict__ gates, int t) {
    int idx = blockIdx.x * blockDim.x + threadIdx.x;      // B*4D
    if (idx >= Bc * 4 * Dc) return;
    int b = idx / (4 * Dc), j = idx % (4 * Dc);
    const float* x = &dec[(b * Tc + t) * Dc];
    float s = 0.f;
    for (int k = 0; k < Dc; ++k) s += x[k] * w_ih[j * (2 * Dc) + k];
    for (int k = 0; k < Dc; ++k) s += cp[b * Dc + k] * w_ih[j * (2 * Dc) + Dc + k];
    for (int k = 0; k < Dc; ++k) s += h[b * Dc + k] * w_hh[j * Dc + k];
    gates[idx] = s;
}

__global__ void lstm_update_kernel(const float* __restrict__ gates, float* __restrict__ cell,
                                   float* __restrict__ h) {
    int idx = blockIdx.x * blockDim.x + threadIdx.x;      // B*D
    if (idx >= Bc * Dc) return;
    int b = idx / Dc, d = idx % Dc;
    const float* g = &gates[b * 4 * Dc];
    float gi = g[d], gf = g[Dc + d], gg = g[2 * Dc + d], go = g[3 * Dc + d];
    float c = sigf(gf) * cell[idx] + sigf(gi) * tanhf(gg);
    cell[idx] = c;
    h[idx] = sigf(go) * tanhf(c);
}

// hw = h @ w_h^T ; also zero the c / cp accumulators for this step
__global__ void hw_zero_kernel(const float* __restrict__ h, const float* __restrict__ w_h,
                               float* __restrict__ hw, float* __restrict__ cbuf,
                               float* __restrict__ cp) {
    int idx = blockIdx.x * blockDim.x + threadIdx.x;      // B*D
    if (idx >= Bc * Dc) return;
    int b = idx / Dc, o = idx % Dc;
    float s = 0.f;
    for (int k = 0; k < Dc; ++k) s += h[b * Dc + k] * w_h[o * Dc + k];
    hw[idx] = s;
    cbuf[idx] = 0.f;
    cp[idx] = 0.f;
}

__global__ void pred_kernel(const float* __restrict__ cbuf, const float* __restrict__ h,
                            const float* __restrict__ w_g, float* __restrict__ out, int t) {
    int idx = blockIdx.x * blockDim.x + threadIdx.x;      // B*D
    if (idx >= Bc * Dc) return;
    int b = idx / Dc, o = idx % Dc;
    float s = 0.f;
    for (int k = 0; k < Dc; ++k) s += cbuf[b * Dc + k] * w_g[o * (2 * Dc) + k];
    for (int k = 0; k < Dc; ++k) s += h[b * Dc + k]    * w_g[o * (2 * Dc) + Dc + k];
    out[(b * Tc + t) * Dc + o] = s;
}

// ---------------------------------------------------------------------------
// Fused attention step: per 16-row tile (same b):
//   t_in = tanh(s@w_s^T + base + hw)   (WMMA GEMM1)
//   e    = t_in @ w_e^T                (WMMA GEMM2)
//   alpha = softmax_row(e); s += alpha; c += alpha*f; cp += alpha*fq
// ---------------------------------------------------------------------------
__global__ __launch_bounds__(256)
void attn_step_kernel(float* __restrict__ s, const float* __restrict__ base,
                      const float* __restrict__ hw, const float* __restrict__ f,
                      const float* __restrict__ fq, const unsigned short* __restrict__ wsb,
                      const unsigned short* __restrict__ web, float* __restrict__ cbuf,
                      float* __restrict__ cp) {
    __shared__ __align__(16) unsigned short As[16 * Dc];
    __shared__ float Es[16 * Dc];
    __shared__ float red[256];
    const int tid = threadIdx.x;
    const int rowbase = blockIdx.x * 16;
    const int b = rowbase >> 9;                       // 512 F-rows per batch
    const int lane = tid & 31, wave = tid >> 5;

    // 1) stage s tile as bf16
    for (int i = tid; i < 16 * Dc; i += 256) As[i] = f2bf(s[rowbase * Dc + i]);
    __syncthreads();

    // 2) GEMM1: s @ w_s^T
    wave_gemm16(As, wsb, Es, lane, wave);
    __syncthreads();

    // 3) t_in = tanh(. + base + hw)  -> bf16 A for GEMM2
    for (int i = tid; i < 16 * Dc; i += 256) {
        int col = i & (Dc - 1);
        float v = Es[i] + base[rowbase * Dc + i] + hw[b * Dc + col];
        As[i] = f2bf(tanhf(v));
    }
    __syncthreads();

    // 4) GEMM2: t_in @ w_e^T
    wave_gemm16(As, web, Es, lane, wave);
    __syncthreads();

    // 5) row softmax over D=256 (16 threads per row)
    {
        int row = tid >> 4, sub = tid & 15;
        float mx = -3.4e38f;
        for (int j = sub; j < Dc; j += 16) mx = fmaxf(mx, Es[row * Dc + j]);
        red[tid] = mx;
        __syncthreads();
        float rmx = red[row * 16];
        for (int j = 1; j < 16; ++j) rmx = fmaxf(rmx, red[row * 16 + j]);
        __syncthreads();
        float sm = 0.f;
        for (int j = sub; j < Dc; j += 16) {
            float v = __expf(Es[row * Dc + j] - rmx);
            Es[row * Dc + j] = v;
            sm += v;
        }
        red[tid] = sm;
        __syncthreads();
        float rs = 0.f;
        for (int j = 0; j < 16; ++j) rs += red[row * 16 + j];
        float inv = 1.0f / rs;
        for (int j = sub; j < Dc; j += 16) Es[row * Dc + j] *= inv;
    }
    __syncthreads();

    // 6) s += alpha ; reduce c / cp (thread t owns output column o = t)
    const int o = tid;
    float ca = 0.f, cpa = 0.f;
    for (int m = 0; m < 16; ++m) {
        int gi = (rowbase + m) * Dc + o;
        float a = Es[m * Dc + o];
        ca  += a * f[gi];
        cpa += a * fq[gi];
        s[gi] += a;
    }
    atomicAdd(&cbuf[b * Dc + o], ca);
    atomicAdd(&cp[b * Dc + o], cpa);
}

// ---------------------------------------------------------------------------
// Host launch
// ---------------------------------------------------------------------------
extern "C" void kernel_launch(void* const* d_in, const int* in_sizes, int n_in,
                              void* d_out, int out_size, void* d_ws, size_t ws_size,
                              hipStream_t stream) {
    (void)in_sizes; (void)n_in; (void)out_size; (void)ws_size;
    const float* f    = (const float*)d_in[0];
    const int*   ids  = (const int*)  d_in[1];
    const float* emb  = (const float*)d_in[2];
    const float* w_e  = (const float*)d_in[3];
    const float* w_h  = (const float*)d_in[4];
    const float* w_f  = (const float*)d_in[5];
    const float* w_q  = (const float*)d_in[6];
    const float* w_s  = (const float*)d_in[7];
    const float* w_g  = (const float*)d_in[8];
    const float* w_ih = (const float*)d_in[9];
    const float* w_hh = (const float*)d_in[10];
    float* out = (float*)d_out;

    char* p = (char*)d_ws;
    auto carve = [&](size_t bytes) -> char* {
        char* r = p;
        p += (bytes + 255) & ~(size_t)255;
        return r;
    };
    float* q     = (float*)carve((size_t)Fc * Dc * 4);
    float* qwq   = (float*)carve((size_t)Fc * Dc * 4);
    float* base  = (float*)carve((size_t)BFc * Dc * 4);
    float* fq    = (float*)carve((size_t)BFc * Dc * 4);
    float* sbuf  = (float*)carve((size_t)BFc * Dc * 4);
    float* dec   = (float*)carve((size_t)Bc * Tc * Dc * 4);
    float* h     = (float*)carve((size_t)Bc * Dc * 4);
    float* cell  = (float*)carve((size_t)Bc * Dc * 4);
    float* cp    = (float*)carve((size_t)Bc * Dc * 4);
    float* cbuf  = (float*)carve((size_t)Bc * Dc * 4);
    float* hw    = (float*)carve((size_t)Bc * Dc * 4);
    float* gates = (float*)carve((size_t)Bc * 4 * Dc * 4);
    unsigned short* wsb = (unsigned short*)carve((size_t)Dc * Dc * 2);
    unsigned short* web = (unsigned short*)carve((size_t)Dc * Dc * 2);
    unsigned short* wfb = (unsigned short*)carve((size_t)Dc * Dc * 2);

    // ---- precompute ----
    pe_kernel<<<(Fc * Dc + 255) / 256, 256, 0, stream>>>(q);
    dec_gather_kernel<<<(Bc * Tc * Dc + 255) / 256, 256, 0, stream>>>(ids, emb, dec);
    cvt_bf16_kernel<<<(Dc * Dc + 255) / 256, 256, 0, stream>>>(w_s, wsb, Dc * Dc);
    cvt_bf16_kernel<<<(Dc * Dc + 255) / 256, 256, 0, stream>>>(w_e, web, Dc * Dc);
    cvt_bf16_kernel<<<(Dc * Dc + 255) / 256, 256, 0, stream>>>(w_f, wfb, Dc * Dc);
    qwq_kernel<<<(Fc * Dc + 255) / 256, 256, 0, stream>>>(q, w_q, qwq);
    base_kernel<<<BFc / 16, 256, 0, stream>>>(f, wfb, qwq, base);
    fq_kernel<<<(Bc * Fc * Dc + 255) / 256, 256, 0, stream>>>(f, q, fq);

    hipMemsetAsync(sbuf, 0, (size_t)BFc * Dc * 4, stream);
    hipMemsetAsync(h,    0, (size_t)Bc * Dc * 4, stream);
    hipMemsetAsync(cell, 0, (size_t)Bc * Dc * 4, stream);
    hipMemsetAsync(cp,   0, (size_t)Bc * Dc * 4, stream);

    // ---- T sequential decode steps ----
    for (int t = 0; t < Tc; ++t) {
        gates_kernel<<<(Bc * 4 * Dc + 255) / 256, 256, 0, stream>>>(dec, cp, h, w_ih, w_hh, gates, t);
        lstm_update_kernel<<<(Bc * Dc + 255) / 256, 256, 0, stream>>>(gates, cell, h);
        hw_zero_kernel<<<(Bc * Dc + 255) / 256, 256, 0, stream>>>(h, w_h, hw, cbuf, cp);
        attn_step_kernel<<<BFc / 16, 256, 0, stream>>>(sbuf, base, hw, f, fq, wsb, web, cbuf, cp);
        pred_kernel<<<(Bc * Dc + 255) / 256, 256, 0, stream>>>(cbuf, h, w_g, out, t);
    }
}
